// ParallelTransformerBlock_43782896615490
// MI455X (gfx1250) — matrix operand order repaired and checked by reading, not compiled, not run on hardware
//
#include <hip/hip_runtime.h>

typedef unsigned short u16;
typedef __attribute__((ext_vector_type(16))) __bf16 v16bf;
typedef __attribute__((ext_vector_type(8)))  float  v8f;
typedef __attribute__((ext_vector_type(8)))  unsigned int v8u;

__device__ __forceinline__ u16 f32_to_bf16(float f) {
    unsigned int u = __builtin_bit_cast(unsigned int, f);
    unsigned int r = (u + 0x7FFFu + ((u >> 16) & 1u)) >> 16;   // round-to-nearest-even
    return (u16)r;
}
__device__ __forceinline__ float bf16_to_f32(u16 h) {
    unsigned int u = ((unsigned int)h) << 16;
    return __builtin_bit_cast(float, u);
}
// A-fragment K offset for VGPR v, lane-half h (16-bit A 16x32 layout, ISA 7.12.2)
__device__ __forceinline__ int a_k0(int v, int h) {
    return ((v < 4) ? 2 * v : 16 + 2 * (v - 4)) + 8 * h;
}

// CDNA5 async global->LDS copy (ASYNCcnt-tracked DMA path, ISA 10.x / 15.18.3).
// lds_dst: generic pointer into __shared__ (low 32 bits == LDS byte offset).
__device__ __forceinline__ void async_ld_b128(void* lds_dst, const void* gsrc) {
    unsigned int       l = (unsigned int)(uintptr_t)lds_dst;
    unsigned long long g = (unsigned long long)(uintptr_t)gsrc;
    asm volatile("global_load_async_to_lds_b128 %0, %1, off" :: "v"(l), "v"(g) : "memory");
}
__device__ __forceinline__ void async_wait0() {
    asm volatile("s_wait_asynccnt 0x0" ::: "memory");
}

// ---------------------------------------------------------------- LayerNorm
__global__ __launch_bounds__(256) void ln_kernel(const float* __restrict__ x,
                                                 const float* __restrict__ gamma,
                                                 u16* __restrict__ xn) {
    __shared__ float red[256];
    const int row = blockIdx.x;                 // 0..4095  (b*2048 + n)
    const float* xr = x + (size_t)row * 2048;
    float s = 0.f, ss = 0.f;
    for (int i = threadIdx.x; i < 2048; i += 256) { float v = xr[i]; s += v; ss += v * v; }
    red[threadIdx.x] = s; __syncthreads();
    for (int o = 128; o > 0; o >>= 1) { if (threadIdx.x < o) red[threadIdx.x] += red[threadIdx.x + o]; __syncthreads(); }
    const float mean = red[0] * (1.f / 2048.f);
    __syncthreads();
    red[threadIdx.x] = ss; __syncthreads();
    for (int o = 128; o > 0; o >>= 1) { if (threadIdx.x < o) red[threadIdx.x] += red[threadIdx.x + o]; __syncthreads(); }
    const float var = red[0] * (1.f / 2048.f) - mean * mean;
    const float inv = rsqrtf(var + 1e-5f);
    for (int i = threadIdx.x; i < 2048; i += 256)
        xn[(size_t)row * 2048 + i] = f32_to_bf16((xr[i] - mean) * inv * gamma[i]);
}

// ---------------------------------------------------------------- fp32 -> bf16
__global__ void cvt_kernel(const float* __restrict__ src, u16* __restrict__ dst, size_t n) {
    size_t i = (size_t)blockIdx.x * blockDim.x + threadIdx.x;
    if (i < n) dst[i] = f32_to_bf16(src[i]);
}

// ---------------------------------------------------------------- bf16 tiled GEMM (WMMA)
// C[M,N] = A[M,K] * B[K,N], A/B bf16 row-major, fp32 accumulate.
// MODE 0: store bf16  MODE 1: store fp32  MODE 2: add fp32
#define BM 128
#define BN 128
#define BKT 32
template <int MODE>
__global__ __launch_bounds__(256) void gemm_bf16(const u16* __restrict__ A,
                                                 const u16* __restrict__ B,
                                                 void* __restrict__ C,
                                                 int M, int N, int K) {
    __shared__ u16 As[BM * BKT];
    __shared__ u16 Bs[BN * BKT];        // stored transposed: Bs[n][k]
    const int tid = threadIdx.x, lane = tid & 31, wid = tid >> 5;
    const int half = lane >> 4, l16 = lane & 15;
    const int wm = (wid & 3) * 32, wn = (wid >> 2) * 64;
    const int bm = blockIdx.y * BM, bn = blockIdx.x * BN;
    v8f acc[2][4] = {};
    for (int kt = 0; kt < K; kt += BKT) {
        {   // A tile: 128 rows x 32 cols, 2 threads/row — async DMA to LDS
            const int row = tid >> 1, seg = (tid & 1) * 16;
            const u16* src = A + (size_t)(bm + row) * K + kt + seg;
            async_ld_b128(&As[row * BKT + seg],     src);
            async_ld_b128(&As[row * BKT + seg + 8], src + 8);
            if (kt + BKT < K) __builtin_prefetch(src + BKT, 0, 1);
        }
        {   // B tile: 32 rows(k) x 128 cols(n), write transposed (VGPR path)
            const int k = tid >> 3, n0 = (tid & 7) * 16;
            const u16* src = B + (size_t)(kt + k) * N + bn + n0;
            uint4 d0 = *(const uint4*)(src);
            uint4 d1 = *(const uint4*)(src + 8);
            const u16* dp = (const u16*)&d0;
#pragma unroll
            for (int i = 0; i < 8; ++i) Bs[(n0 + i) * BKT + k] = dp[i];
            dp = (const u16*)&d1;
#pragma unroll
            for (int i = 0; i < 8; ++i) Bs[(n0 + 8 + i) * BKT + k] = dp[i];
            if (kt + BKT < K) __builtin_prefetch(src + (size_t)BKT * N, 0, 1);
        }
        async_wait0();
        __syncthreads();
        v16bf af[2], bfr[4];
#pragma unroll
        for (int mt = 0; mt < 2; ++mt) {
            const int m = wm + mt * 16 + l16;
            v8u t;
#pragma unroll
            for (int v = 0; v < 8; ++v)
                t[v] = *(const unsigned int*)&As[m * BKT + a_k0(v, half)];
            af[mt] = __builtin_bit_cast(v16bf, t);
        }
#pragma unroll
        for (int nt = 0; nt < 4; ++nt) {
            const int n = wn + nt * 16 + l16;
            v8u t;
#pragma unroll
            for (int v = 0; v < 8; ++v)
                t[v] = *(const unsigned int*)&Bs[n * BKT + a_k0(v, half)];
            bfr[nt] = __builtin_bit_cast(v16bf, t);
        }
#pragma unroll
        for (int mt = 0; mt < 2; ++mt)
#pragma unroll
            for (int nt = 0; nt < 4; ++nt)
                acc[mt][nt] = __builtin_amdgcn_wmma_f32_16x16x32_bf16(
                    false, af[mt], false, bfr[nt], (short)0, acc[mt][nt], false, false);
        __syncthreads();
    }
#pragma unroll
    for (int mt = 0; mt < 2; ++mt)
#pragma unroll
        for (int nt = 0; nt < 4; ++nt)
#pragma unroll
            for (int r = 0; r < 8; ++r) {
                const int row = bm + wm + mt * 16 + r + 8 * half;
                const int col = bn + wn + nt * 16 + l16;
                const float v = acc[mt][nt][r];
                if (MODE == 0)      ((u16*)C)[(size_t)row * N + col] = f32_to_bf16(v);
                else if (MODE == 1) ((float*)C)[(size_t)row * N + col] = v;
                else                ((float*)C)[(size_t)row * N + col] += v;
            }
}

// ---------------------------------------------------------------- RoPE (xpos) + split q/k/v
__global__ __launch_bounds__(256) void rope_kernel(const u16* __restrict__ proj,
                                                   u16* __restrict__ q,
                                                   u16* __restrict__ k,
                                                   u16* __restrict__ v) {
    const int n = blockIdx.x & 2047, b = blockIdx.x >> 11;
    const u16* pr = proj + (size_t)(b * 2048 + n) * 18688;
    const float power = (float)(n - 1024) * (1.f / 512.f);
    for (int idx = threadIdx.x; idx < 2048; idx += 256) {
        const int hh = idx >> 7, d = idx & 127, i = d & 63;
        const float inv_freq = powf(10000.f, -(2.f * i) / 128.f);
        const float ang = (float)n * inv_freq;
        const float sv = (2.f * i + 51.2f) / 179.2f;
        const float s = powf(sv, power);
        const float xv = bf16_to_f32(pr[idx]);
        const float partner = bf16_to_f32(pr[(d < 64) ? idx + 64 : idx - 64]);
        const float sign = (d < 64) ? -1.f : 1.f;
        const float o = xv * cosf(ang) * s + sign * partner * sinf(ang) * s;
        q[(((size_t)(b * 16 + hh)) * 2048 + n) * 128 + d] = f32_to_bf16(o);
    }
    if (threadIdx.x < 128) {
        const int d = threadIdx.x, i = d & 63;
        const float inv_freq = powf(10000.f, -(2.f * i) / 128.f);
        const float ang = (float)n * inv_freq;
        const float sv = (2.f * i + 51.2f) / 179.2f;
        const float s = 1.f / powf(sv, power);
        const float xv = bf16_to_f32(pr[2048 + d]);
        const float partner = bf16_to_f32(pr[2048 + ((d < 64) ? d + 64 : d - 64)]);
        const float sign = (d < 64) ? -1.f : 1.f;
        k[((size_t)(b * 2048 + n)) * 128 + d] =
            f32_to_bf16(xv * cosf(ang) * s + sign * partner * sinf(ang) * s);
        v[((size_t)(b * 2048 + n)) * 128 + d] = pr[2176 + d];
    }
}

// ---------------------------------------------------------------- swiglu ff activation
__global__ void ffact_kernel(const u16* __restrict__ proj, u16* __restrict__ ffact) {
    const size_t i = (size_t)blockIdx.x * 256 + threadIdx.x;    // 4096*8192 exact
    const size_t row = i >> 13; const int j = (int)(i & 8191);
    const u16* pr = proj + row * 18688 + 2304;
    const float xv = bf16_to_f32(pr[j]);
    const float g  = bf16_to_f32(pr[8192 + j]);
    ffact[i] = f32_to_bf16(g / (1.f + expf(-g)) * xv);
}

// ---------------------------------------------------------------- flash attention (MQA, causal)
// grid (n/128, H, B), 256 threads = 8 waves; each wave owns 16 query rows.
__global__ __launch_bounds__(256) void attn_kernel(const u16* __restrict__ q,
                                                   const u16* __restrict__ k,
                                                   const u16* __restrict__ v,
                                                   u16* __restrict__ ctx) {
    __shared__ u16 Ks[128 * 128];
    __shared__ u16 Vs[128 * 128];
    __shared__ u16 Ps[8][16 * 128];
    const int tid = threadIdx.x, lane = tid & 31, wid = tid >> 5;
    const int half = lane >> 4, l16 = lane & 15;
    const int mblock = blockIdx.x, h = blockIdx.y, b = blockIdx.z;
    const int Dh = 128;
    const u16* qb = q + ((size_t)(b * 16 + h) * 2048) * Dh;
    const u16* kb = k + (size_t)b * 2048 * Dh;
    const u16* vb = v + (size_t)b * 2048 * Dh;
    const int m0 = mblock * 128 + wid * 16;

    v16bf qf[4];
#pragma unroll
    for (int kc = 0; kc < 4; ++kc) {
        v8u t;
#pragma unroll
        for (int vv = 0; vv < 8; ++vv)
            t[vv] = *(const unsigned int*)(qb + (size_t)(m0 + l16) * Dh + kc * 32 + a_k0(vv, half));
        qf[kc] = __builtin_bit_cast(v16bf, t);
    }
    float rmax[8], rsum[8];
    v8f o[8] = {};
#pragma unroll
    for (int r = 0; r < 8; ++r) { rmax[r] = -1e30f; rsum[r] = 0.f; }
    const float sc = 0.08838834764831845f;   // 128^-0.5

    for (int jt = 0; jt <= mblock; ++jt) {
        {   // stage K,V tiles (128 keys x 128 dh) via async DMA to LDS
            const int row = tid >> 1, seg = (tid & 1) * 64;
            const u16* ks = kb + (size_t)(jt * 128 + row) * Dh + seg;
            const u16* vs = vb + (size_t)(jt * 128 + row) * Dh + seg;
#pragma unroll
            for (int i = 0; i < 4; ++i) {
                async_ld_b128(&Ks[row * 128 + seg + i * 16], ks + i * 16);
                async_ld_b128(&Vs[row * 128 + seg + i * 16], vs + i * 16);
            }
        }
        async_wait0();
        __syncthreads();
        v8f st[8];
#pragma unroll
        for (int nt = 0; nt < 8; ++nt) {
            v8f s = {};
#pragma unroll
            for (int kc = 0; kc < 4; ++kc) {
                v8u t;   // B-frag of K^T: n=key, k=d (contiguous in d -> 32-bit ds reads)
#pragma unroll
                for (int vv = 0; vv < 8; ++vv)
                    t[vv] = *(const unsigned int*)&Ks[(nt * 16 + l16) * 128 + kc * 32 + a_k0(vv, half)];
                v16bf kf = __builtin_bit_cast(v16bf, t);
                s = __builtin_amdgcn_wmma_f32_16x16x32_bf16(false, qf[kc], false, kf, (short)0, s, false, false);
            }
            const int keyg = jt * 128 + nt * 16 + l16;
#pragma unroll
            for (int r = 0; r < 8; ++r) {
                const int rowg = m0 + r + 8 * half;
                s[r] = (keyg > rowg) ? -1e30f : s[r] * sc;
            }
            st[nt] = s;
        }
        // online softmax: each row lives in one VGPR across a 16-lane group
#pragma unroll
        for (int r = 0; r < 8; ++r) {
            float tmax = -1e30f;
#pragma unroll
            for (int nt = 0; nt < 8; ++nt) tmax = fmaxf(tmax, st[nt][r]);
#pragma unroll
            for (int off = 1; off < 16; off <<= 1) tmax = fmaxf(tmax, __shfl_xor(tmax, off, 32));
            const float newm = fmaxf(rmax[r], tmax);
            const float corr = expf(rmax[r] - newm);
            rmax[r] = newm;
            float ps = 0.f;
#pragma unroll
            for (int nt = 0; nt < 8; ++nt) { const float p = expf(st[nt][r] - newm); st[nt][r] = p; ps += p; }
#pragma unroll
            for (int off = 1; off < 16; off <<= 1) ps += __shfl_xor(ps, off, 32);
            rsum[r] = rsum[r] * corr + ps;
#pragma unroll
            for (int nt = 0; nt < 8; ++nt) o[nt][r] *= corr;
        }
        // spill P (C-layout) to per-wave LDS, reload as A-fragments
#pragma unroll
        for (int nt = 0; nt < 8; ++nt)
#pragma unroll
            for (int r = 0; r < 8; ++r)
                Ps[wid][(r + 8 * half) * 128 + nt * 16 + l16] = f32_to_bf16(st[nt][r]);
#pragma unroll
        for (int kc = 0; kc < 4; ++kc) {
            v8u t;
#pragma unroll
            for (int vv = 0; vv < 8; ++vv)
                t[vv] = *(const unsigned int*)&Ps[wid][l16 * 128 + kc * 32 + a_k0(vv, half)];
            v16bf pf = __builtin_bit_cast(v16bf, t);
#pragma unroll
            for (int nt = 0; nt < 8; ++nt) {
                v8u tv;  // B-frag of V: k=key (stride 128 in LDS), n=d
#pragma unroll
                for (int vv = 0; vv < 8; ++vv) {
                    const int kk = kc * 32 + a_k0(vv, half);
                    const unsigned int lo = Vs[kk * 128 + nt * 16 + l16];
                    const unsigned int hi = Vs[(kk + 1) * 128 + nt * 16 + l16];
                    tv[vv] = lo | (hi << 16);
                }
                v16bf vf = __builtin_bit_cast(v16bf, tv);
                o[nt] = __builtin_amdgcn_wmma_f32_16x16x32_bf16(false, pf, false, vf, (short)0, o[nt], false, false);
            }
        }
        __syncthreads();
    }
#pragma unroll
    for (int nt = 0; nt < 8; ++nt)
#pragma unroll
        for (int r = 0; r < 8; ++r) {
            const int rowg = m0 + r + 8 * half;
            const int d = nt * 16 + l16;
            ctx[((size_t)(b * 2048 + rowg)) * 2048 + h * 128 + d] = f32_to_bf16(o[nt][r] / rsum[r]);
        }
}

// ---------------------------------------------------------------- host
extern "C" void kernel_launch(void* const* d_in, const int* in_sizes, int n_in,
                              void* d_out, int out_size, void* d_ws, size_t ws_size,
                              hipStream_t stream) {
    const float* x      = (const float*)d_in[0];
    const float* gamma  = (const float*)d_in[1];
    const float* Wfused = (const float*)d_in[2];
    const float* Wao    = (const float*)d_in[3];
    const float* Wff    = (const float*)d_in[4];
    float* out = (float*)d_out;

    size_t off = 0;
    auto alloc = [&](size_t bytes) {
        char* p = (char*)d_ws + off;
        off += (bytes + 255) & ~(size_t)255;
        return (u16*)p;
    };
    u16* xn    = alloc((size_t)4096 * 2048 * 2);
    u16* WfB   = alloc((size_t)2048 * 18688 * 2);
    u16* WaoB  = alloc((size_t)2048 * 2048 * 2);
    u16* WffB  = alloc((size_t)8192 * 2048 * 2);
    u16* proj  = alloc((size_t)4096 * 18688 * 2);
    u16* qbuf  = alloc((size_t)2 * 16 * 2048 * 128 * 2);
    u16* kbuf  = alloc((size_t)2 * 2048 * 128 * 2);
    u16* vbuf  = alloc((size_t)2 * 2048 * 128 * 2);
    u16* ffa   = alloc((size_t)4096 * 8192 * 2);
    u16* ctx   = alloc((size_t)4096 * 2048 * 2);

    // 1. weight conversion to bf16
    {
        size_t n1 = (size_t)2048 * 18688, n2 = (size_t)2048 * 2048, n3 = (size_t)8192 * 2048;
        cvt_kernel<<<dim3((unsigned)((n1 + 255) / 256)), 256, 0, stream>>>(Wfused, WfB, n1);
        cvt_kernel<<<dim3((unsigned)((n2 + 255) / 256)), 256, 0, stream>>>(Wao,    WaoB, n2);
        cvt_kernel<<<dim3((unsigned)((n3 + 255) / 256)), 256, 0, stream>>>(Wff,    WffB, n3);
    }
    // 2. LayerNorm
    ln_kernel<<<4096, 256, 0, stream>>>(x, gamma, xn);
    // 3. fused projection: proj = xn @ W_fused   (4096 x 18688 x 2048)
    gemm_bf16<0><<<dim3(18688 / BN, 4096 / BM), 256, 0, stream>>>(xn, WfB, proj, 4096, 18688, 2048);
    // 4. RoPE + q/k/v split
    rope_kernel<<<4096, 256, 0, stream>>>(proj, qbuf, kbuf, vbuf);
    // 5. swiglu ff activation
    ffact_kernel<<<(unsigned)(((size_t)4096 * 8192) / 256), 256, 0, stream>>>(proj, ffa);
    // 6. causal MQA flash attention
    attn_kernel<<<dim3(2048 / 128, 16, 2), 256, 0, stream>>>(qbuf, kbuf, vbuf, ctx);
    // 7. out = ctx @ W_attn_out (fp32 store)
    gemm_bf16<1><<<dim3(2048 / BN, 4096 / BM), 256, 0, stream>>>(ctx, WaoB, out, 4096, 2048, 2048);
    // 8. out += ffact @ W_ff_out (fp32 add)
    gemm_bf16<2><<<dim3(2048 / BN, 4096 / BM), 256, 0, stream>>>(ffa, WffB, out, 4096, 2048, 8192);
    (void)in_sizes; (void)n_in; (void)out_size; (void)ws_size;
}
